// DecoderLayer_2147483648548
// MI455X (gfx1250) — matrix operand order repaired and checked by reading, not compiled, not run on hardware
//
#include <hip/hip_runtime.h>

#define D_MODEL 1024
#define D_FF    4096
#define N_HEAD  16
#define BATCH   4
#define SEQ_L   1024
#define SEQ_S   1024
#define D_HEAD  64

typedef __attribute__((ext_vector_type(16))) __bf16 v16bf;
typedef __attribute__((ext_vector_type(8)))  float  v8f;

struct U32x8 { uint4 lo, hi; };

static __device__ inline v16bf frag_cast(uint4 lo, uint4 hi) {
  U32x8 t; t.lo = lo; t.hi = hi;
  return __builtin_bit_cast(v16bf, t);
}

// f32 -> bf16 round-to-nearest-even
static __device__ inline unsigned short f2bf(float f) {
  unsigned int u = __float_as_uint(f);
  u += 0x7FFFu + ((u >> 16) & 1u);
  return (unsigned short)(u >> 16);
}

// CDNA5 async global->LDS copy (ASYNCcnt-tracked, no VGPR round trip).
// ldsOff = byte offset in LDS (low 32 bits of flat LDS pointer),
// gaddr  = per-lane 64-bit global address. 16 bytes per lane.
static __device__ inline void async_ld_b128(unsigned ldsOff, unsigned long long gaddr) {
  asm volatile("global_load_async_to_lds_b128 %0, %1, off"
               :: "v"(ldsOff), "v"(gaddr) : "memory");
}
static __device__ inline void wait_async0() {
  asm volatile("s_wait_asynccnt 0x0" ::: "memory");
}

// ---------------------------------------------------------------------------
// GEMM: out[z,m,n] = act(alpha * sum_k A[m,k]*B(k,n) + bias[n]) (+ resid[m,n])
//   A[m,k]  at A + z*aZ + m*lda + k                     (bf16)
//   B(k,n)  at Bp + z*bZ + n*bldn + k   (bKContig==1, weight W[n,k] layout)
//           at Bp + z*bZ + k*bldk + n   (bKContig==0, V[k,n] layout)
// Tile 128x64x64, 8 wave32s, each wave 32x32, 8x v_wmma_f32_16x16x32_bf16
// per stage; A/B staged with global_load_async_to_lds_b128.
// ---------------------------------------------------------------------------
#define BM 128
#define BN 64
#define BK 64
#define LDSP (BK + 8)   // padded LDS row (ushorts): 144B, 16B-aligned

__global__ __launch_bounds__(256, 2)
void gemm_bf16_wmma(const unsigned short* __restrict__ A, long lda, long aZ,
                    const unsigned short* __restrict__ Bp, long bldk, long bldn,
                    long bZ, int bKContig,
                    const float* __restrict__ bias,
                    const float* __restrict__ resid,
                    float* __restrict__ outF, unsigned short* __restrict__ outB,
                    long ldo, long oZ,
                    int M, int N, int K, float alpha, int act)
{
  __shared__ unsigned short As[BM * LDSP];
  __shared__ unsigned short Bs[BN * LDSP];
  const int t    = threadIdx.x;
  const int lane = t & 31;
  const int wave = t >> 5;
  const int hl   = lane >> 4;     // lane half (0/1)
  const int r    = lane & 15;
  const int wm   = (wave & 3) * 32;
  const int wn   = (wave >> 2) * 32;
  const long z   = blockIdx.z;
  const unsigned short* Ab = A + z * aZ + (long)blockIdx.y * BM * lda;
  const unsigned short* Bb = Bp + z * bZ;
  const long bn0 = (long)blockIdx.x * BN;
  (void)M; (void)N;

  const unsigned asBase = (unsigned)(unsigned long long)(void*)As;
  const unsigned bsBase = (unsigned)(unsigned long long)(void*)Bs;

  const v8f vzero = {0.f,0.f,0.f,0.f,0.f,0.f,0.f,0.f};
  v8f acc[2][2];
  acc[0][0] = vzero; acc[0][1] = vzero; acc[1][0] = vzero; acc[1][1] = vzero;

  for (int k0 = 0; k0 < K; k0 += BK) {
    // ---- stage A tile (128x64): 1024 16B chunks, 4 async copies/thread
    #pragma unroll
    for (int i = 0; i < 4; ++i) {
      int ci  = t + i * 256;
      int row = ci >> 3;          // 8 chunks per 64-wide row
      int kc  = (ci & 7) * 8;
      async_ld_b128(asBase + (unsigned)(row * LDSP + kc) * 2u,
                    (unsigned long long)(Ab + (long)row * lda + (k0 + kc)));
    }
    // ---- stage B tile (64x64) as Bs[n][k]
    if (bKContig) {
      #pragma unroll
      for (int i = 0; i < 2; ++i) {
        int ci  = t + i * 256;
        int row = ci >> 3;
        int kc  = (ci & 7) * 8;
        async_ld_b128(bsBase + (unsigned)(row * LDSP + kc) * 2u,
                      (unsigned long long)(Bb + (bn0 + row) * bldn + (k0 + kc)));
      }
    } else {
      // B(k,n) = Bb[k*bldk + n]: vector-load n-contiguous, scatter-transpose
      #pragma unroll
      for (int i = 0; i < 2; ++i) {
        int ci = t + i * 256;
        int kk = ci >> 3;          // 0..63
        int nc = (ci & 7) * 8;     // 0..56
        uint4 v = *(const uint4*)(Bb + (long)(k0 + kk) * bldk + (bn0 + nc));
        const unsigned short* pv = (const unsigned short*)&v;
        #pragma unroll
        for (int j = 0; j < 8; ++j)
          Bs[(nc + j) * LDSP + kk] = pv[j];
      }
    }
    wait_async0();
    __syncthreads();

    // ---- two 32-deep WMMA stages per buffer (ISA 7.12.2 16-bit layouts)
    #pragma unroll
    for (int ks = 0; ks < BK; ks += 32) {
      v16bf afrag[2], bfrag[2];
      #pragma unroll
      for (int ti = 0; ti < 2; ++ti) {
        // lane half 0: K ks+0..7 / ks+16..23 ; half 1: K ks+8..15 / ks+24..31
        const unsigned short* ap = &As[(wm + ti * 16 + r) * LDSP + ks + hl * 8];
        afrag[ti] = frag_cast(*(const uint4*)ap, *(const uint4*)(ap + 16));
      }
      #pragma unroll
      for (int tj = 0; tj < 2; ++tj) {
        // lane half 0: K ks+0..7 / ks+16..23 ; half 1: K ks+8..15 / ks+24..31 (column = r)
        const unsigned short* bp = &Bs[(wn + tj * 16 + r) * LDSP + ks + hl * 8];
        bfrag[tj] = frag_cast(*(const uint4*)bp, *(const uint4*)(bp + 16));
      }
      #pragma unroll
      for (int ti = 0; ti < 2; ++ti)
        #pragma unroll
        for (int tj = 0; tj < 2; ++tj)
          acc[ti][tj] = __builtin_amdgcn_wmma_f32_16x16x32_bf16(
              false, afrag[ti], false, bfrag[tj], (short)0, acc[ti][tj],
              false, false);
    }
    __syncthreads();
  }

  // ---- epilogue: C/D layout: element i of v8f -> row = i + hl*8, col = r
  const long mBase = (long)blockIdx.y * BM + wm + hl * 8;
  #pragma unroll
  for (int ti = 0; ti < 2; ++ti) {
    #pragma unroll
    for (int tj = 0; tj < 2; ++tj) {
      const long col = bn0 + wn + tj * 16 + r;
      const float bv = bias ? bias[col] : 0.0f;
      #pragma unroll
      for (int i = 0; i < 8; ++i) {
        const long row = mBase + ti * 16 + i;
        float v = acc[ti][tj][i] * alpha + bv;
        if (act == 1) v = 0.5f * v * (1.0f + erff(v * 0.70710678118654752f));
        if (resid) v += resid[row * ldo + col];
        const long o = z * oZ + row * ldo + col;
        if (outF) outF[o] = v;
        if (outB) outB[o] = f2bf(v);
      }
    }
  }
}

// ---------------------------------------------------------------------------
// LayerNorm over D=1024: 256 threads, 4 elems/thread; writes f32 + bf16.
// ---------------------------------------------------------------------------
__global__ __launch_bounds__(256)
void ln_kernel(const float* __restrict__ in, const float* __restrict__ w,
               const float* __restrict__ b, float* __restrict__ outF,
               unsigned short* __restrict__ outB, int D)
{
  __shared__ float red[256];
  const long row = blockIdx.x;
  const int t = threadIdx.x;
  float4 v = ((const float4*)(in + row * D))[t];
  red[t] = v.x + v.y + v.z + v.w;
  __syncthreads();
  for (int o = 128; o > 0; o >>= 1) { if (t < o) red[t] += red[t + o]; __syncthreads(); }
  const float mean = red[0] * (1.0f / D);
  __syncthreads();
  float4 d;
  d.x = v.x - mean; d.y = v.y - mean; d.z = v.z - mean; d.w = v.w - mean;
  red[t] = d.x*d.x + d.y*d.y + d.z*d.z + d.w*d.w;
  __syncthreads();
  for (int o = 128; o > 0; o >>= 1) { if (t < o) red[t] += red[t + o]; __syncthreads(); }
  const float rstd = rsqrtf(red[0] * (1.0f / D) + 1e-5f);
  float4 wv = ((const float4*)w)[t];
  float4 bv = ((const float4*)b)[t];
  float4 y;
  y.x = d.x * rstd * wv.x + bv.x;
  y.y = d.y * rstd * wv.y + bv.y;
  y.z = d.z * rstd * wv.z + bv.z;
  y.w = d.w * rstd * wv.w + bv.w;
  ((float4*)(outF + row * D))[t] = y;
  ushort4 yb; yb.x = f2bf(y.x); yb.y = f2bf(y.y); yb.z = f2bf(y.z); yb.w = f2bf(y.w);
  ((ushort4*)(outB + row * D))[t] = yb;
}

// ---------------------------------------------------------------------------
// Softmax over S for one query row l, looping all H heads; writes bf16 probs
// and the head-averaged weights (no atomics).
// ---------------------------------------------------------------------------
__global__ __launch_bounds__(256)
void softmax_attn_kernel(const float* __restrict__ scores,
                         unsigned short* __restrict__ attnb,
                         float* __restrict__ meanout, int L, int S, int H)
{
  __shared__ float red[256];
  const long l = blockIdx.x;
  const int t = threadIdx.x;
  const float invH = 1.0f / H;
  float m0 = 0.f, m1 = 0.f, m2 = 0.f, m3 = 0.f;
  for (int h = 0; h < H; ++h) {
    const float* p = scores + ((long)h * L + l) * S;
    float4 v = ((const float4*)p)[t];
    red[t] = fmaxf(fmaxf(v.x, v.y), fmaxf(v.z, v.w));
    __syncthreads();
    for (int o = 128; o > 0; o >>= 1) { if (t < o) red[t] = fmaxf(red[t], red[t + o]); __syncthreads(); }
    const float rmax = red[0];
    __syncthreads();
    float e0 = __expf(v.x - rmax), e1 = __expf(v.y - rmax);
    float e2 = __expf(v.z - rmax), e3 = __expf(v.w - rmax);
    red[t] = e0 + e1 + e2 + e3;
    __syncthreads();
    for (int o = 128; o > 0; o >>= 1) { if (t < o) red[t] += red[t + o]; __syncthreads(); }
    const float inv = 1.0f / red[0];
    __syncthreads();
    e0 *= inv; e1 *= inv; e2 *= inv; e3 *= inv;
    ushort4 ob; ob.x = f2bf(e0); ob.y = f2bf(e1); ob.z = f2bf(e2); ob.w = f2bf(e3);
    ((ushort4*)(attnb + ((long)h * L + l) * S))[t] = ob;
    m0 += e0 * invH; m1 += e1 * invH; m2 += e2 * invH; m3 += e3 * invH;
  }
  float4 mo; mo.x = m0; mo.y = m1; mo.z = m2; mo.w = m3;
  ((float4*)(meanout + l * S))[t] = mo;
}

// ---------------------------------------------------------------------------
__global__ __launch_bounds__(256)
void f32_to_bf16_kernel(const float* __restrict__ in,
                        unsigned short* __restrict__ out, long n)
{
  long i = ((long)blockIdx.x * 256 + threadIdx.x) * 4;
  if (i >= n) return;
  float4 v = *(const float4*)(in + i);
  ushort4 o; o.x = f2bf(v.x); o.y = f2bf(v.y); o.z = f2bf(v.z); o.w = f2bf(v.w);
  *(ushort4*)(out + i) = o;
}

// ---------------------------------------------------------------------------
extern "C" void kernel_launch(void* const* d_in, const int* in_sizes, int n_in,
                              void* d_out, int out_size, void* d_ws, size_t ws_size,
                              hipStream_t stream)
{
  (void)in_sizes; (void)n_in; (void)out_size; (void)ws_size;
  const float* x        = (const float*)d_in[0];
  const float* xa       = (const float*)d_in[1];
  const float* sa_in_w  = (const float*)d_in[2];
  const float* sa_in_b  = (const float*)d_in[3];
  const float* sa_out_w = (const float*)d_in[4];
  const float* sa_out_b = (const float*)d_in[5];
  const float* ln1_w    = (const float*)d_in[6];
  const float* ln1_b    = (const float*)d_in[7];
  const float* ca_in_w  = (const float*)d_in[8];
  const float* ca_in_b  = (const float*)d_in[9];
  const float* ca_out_w = (const float*)d_in[10];
  const float* ca_out_b = (const float*)d_in[11];
  const float* ln2_w    = (const float*)d_in[12];
  const float* ln2_b    = (const float*)d_in[13];
  const float* ff_w1    = (const float*)d_in[14];
  const float* ff_b1    = (const float*)d_in[15];
  const float* ff_w2    = (const float*)d_in[16];
  const float* ff_b2    = (const float*)d_in[17];
  const float* ln3_w    = (const float*)d_in[18];
  const float* ln3_b    = (const float*)d_in[19];

  const long Mr = (long)BATCH * SEQ_L;   // 4096 rows
  const long D3 = 3 * D_MODEL;           // 3072

  float* out_x     = (float*)d_out;
  float* out_self  = out_x + Mr * D_MODEL;
  float* out_cross = out_self + (long)BATCH * SEQ_L * SEQ_S;

  size_t off = 0;
  auto carve = [&](size_t bytes) -> void* {
    void* p = (char*)d_ws + off;
    off += (bytes + 255) & ~(size_t)255;
    return p;
  };
  float*          lnF    = (float*)carve((size_t)Mr * D_MODEL * 4);
  unsigned short* lnB    = (unsigned short*)carve((size_t)Mr * D_MODEL * 2);
  float*          resid  = (float*)carve((size_t)Mr * D_MODEL * 4);
  unsigned short* qkv    = (unsigned short*)carve((size_t)Mr * D3 * 2);
  float*          scores = (float*)carve((size_t)N_HEAD * SEQ_L * SEQ_S * 4);
  unsigned short* attnb  = (unsigned short*)carve((size_t)N_HEAD * SEQ_L * SEQ_S * 2);
  unsigned short* ctx    = (unsigned short*)carve((size_t)Mr * D_MODEL * 2);
  unsigned short* wb     = (unsigned short*)carve((size_t)D_FF * D_MODEL * 2);
  unsigned short* xab    = (unsigned short*)carve((size_t)Mr * D_MODEL * 2);
  unsigned short* ffact  = (unsigned short*)carve((size_t)Mr * D_FF * 2);

  auto conv = [&](const float* src, unsigned short* dst, long n) {
    f32_to_bf16_kernel<<<dim3((unsigned)(n / 1024)), dim3(256), 0, stream>>>(src, dst, n);
  };
  auto gemm = [&](const unsigned short* Ap, long lda, long aZ,
                  const unsigned short* Bpp, long bldk, long bldn, long bZ, int bKC,
                  const float* bias, const float* res,
                  float* oF, unsigned short* oB, long ldo, long oZ,
                  int Mq, int Nq, int Kq, float alpha, int act, int Z) {
    dim3 g((unsigned)(Nq / BN), (unsigned)(Mq / BM), (unsigned)Z);
    gemm_bf16_wmma<<<g, dim3(256), 0, stream>>>(Ap, lda, aZ, Bpp, bldk, bldn, bZ, bKC,
                                                bias, res, oF, oB, ldo, oZ,
                                                Mq, Nq, Kq, alpha, act);
  };
  auto attention = [&](float* out_mean) {
    for (int b = 0; b < BATCH; ++b) {
      const unsigned short* qb = qkv + (long)b * SEQ_L * D3;
      // scores[h,l,s] = (1/sqrt(64)) * q[l,:] . k[s,:]
      gemm(qb, D3, D_HEAD, qb + D_MODEL, 1, D3, D_HEAD, 1,
           nullptr, nullptr, scores, nullptr, SEQ_S, (long)SEQ_L * SEQ_S,
           SEQ_L, SEQ_S, D_HEAD, 0.125f, 0, N_HEAD);
      softmax_attn_kernel<<<dim3(SEQ_L), dim3(256), 0, stream>>>(
          scores, attnb, out_mean + (long)b * SEQ_L * SEQ_S, SEQ_L, SEQ_S, N_HEAD);
      // ctx[l, h*64+d] = sum_s attn[h,l,s] * v[s, h*64+d]
      gemm(attnb, SEQ_S, (long)SEQ_L * SEQ_S, qb + 2 * D_MODEL, D3, 1, D_HEAD, 0,
           nullptr, nullptr, nullptr, ctx + (long)b * SEQ_L * D_MODEL, D_MODEL, D_HEAD,
           SEQ_L, D_HEAD, SEQ_S, 1.0f, 0, N_HEAD);
    }
  };

  // x = LN1(x)  (reference overwrites x; lnF is the residual base)
  ln_kernel<<<dim3((unsigned)Mr), dim3(256), 0, stream>>>(x, ln1_w, ln1_b, lnF, lnB, D_MODEL);
  // self-attention QKV projection
  conv(sa_in_w, wb, D3 * D_MODEL);
  gemm(lnB, D_MODEL, 0, wb, 1, D_MODEL, 0, 1, sa_in_b, nullptr,
       nullptr, qkv, D3, 0, (int)Mr, (int)D3, D_MODEL, 1.0f, 0, 1);
  attention(out_self);
  // x1 = LN1(x) + attn_out @ Wo + bo
  conv(sa_out_w, wb, (long)D_MODEL * D_MODEL);
  gemm(ctx, D_MODEL, 0, wb, 1, D_MODEL, 0, 1, sa_out_b, lnF,
       resid, nullptr, D_MODEL, 0, (int)Mr, D_MODEL, D_MODEL, 1.0f, 0, 1);
  // x2 = LN2(x1)
  ln_kernel<<<dim3((unsigned)Mr), dim3(256), 0, stream>>>(resid, ln2_w, ln2_b, lnF, lnB, D_MODEL);
  // cross-attention: q from x2, k/v from xa
  conv(ca_in_w, wb, (long)D_MODEL * D_MODEL);
  gemm(lnB, D_MODEL, 0, wb, 1, D_MODEL, 0, 1, ca_in_b, nullptr,
       nullptr, qkv, D3, 0, (int)Mr, D_MODEL, D_MODEL, 1.0f, 0, 1);
  conv(xa, xab, Mr * D_MODEL);
  conv(ca_in_w + (long)D_MODEL * D_MODEL, wb, (long)2 * D_MODEL * D_MODEL);
  gemm(xab, D_MODEL, 0, wb, 1, D_MODEL, 0, 1, ca_in_b + D_MODEL, nullptr,
       nullptr, qkv + D_MODEL, D3, 0, (int)Mr, 2 * D_MODEL, D_MODEL, 1.0f, 0, 1);
  attention(out_cross);
  // x3 = x2 + cross_out @ Wo + bo
  conv(ca_out_w, wb, (long)D_MODEL * D_MODEL);
  gemm(ctx, D_MODEL, 0, wb, 1, D_MODEL, 0, 1, ca_out_b, lnF,
       resid, nullptr, D_MODEL, 0, (int)Mr, D_MODEL, D_MODEL, 1.0f, 0, 1);
  // h = LN3(x3)
  ln_kernel<<<dim3((unsigned)Mr), dim3(256), 0, stream>>>(resid, ln3_w, ln3_b, lnF, lnB, D_MODEL);
  // FFN: gelu(h @ W1^T + b1) @ W2^T + b2 + x3
  conv(ff_w1, wb, (long)D_FF * D_MODEL);
  gemm(lnB, D_MODEL, 0, wb, 1, D_MODEL, 0, 1, ff_b1, nullptr,
       nullptr, ffact, D_FF, 0, (int)Mr, D_FF, D_MODEL, 1.0f, 1, 1);
  conv(ff_w2, wb, (long)D_MODEL * D_FF);
  gemm(ffact, D_FF, 0, wb, 1, D_FF, 0, 1, ff_b2, resid,
       out_x, nullptr, D_MODEL, 0, (int)Mr, D_MODEL, D_FF, 1.0f, 0, 1);
}
